// LocalPoolDGCNN_37374805410428
// MI455X (gfx1250) — matrix operand order repaired and checked by reading, not compiled, not run on hardware
//
#include <hip/hip_runtime.h>
#include <hip/hip_bf16.h>

// ---------------------------------------------------------------------------
// LocalPoolDGCNN for MI455X (gfx1250, wave32).
// Heavy GEMMs (edge convs 2-4, final projection) on v_wmma_f32_16x16x32_bf16
// with f32 accumulation. Weights pre-converted to bf16 once; edge-conv stages
// them in LDS (conflict-free padded layout) via async-to-LDS copies when the
// toolchain exposes them. KNN = register top-16 over LDS candidate chunks.
// ---------------------------------------------------------------------------

#define BATCH 4
#define NPTS  4096
#define KNN_K 16
#define HDIM  128
#define WLD   264   // padded LDS row stride (ushorts): 132 dwords -> bank-conflict-free

typedef __attribute__((ext_vector_type(16))) __bf16 v16bf;
typedef __attribute__((ext_vector_type(8)))  float  v8f;
typedef __attribute__((ext_vector_type(4)))  int    v4i;

union ABu {
    v16bf          v;
    __bf16         h[16];
    unsigned short u[16];
    uint4          q[2];
};

#if __has_builtin(__builtin_amdgcn_global_load_async_to_lds_b128) && \
    __has_builtin(__builtin_amdgcn_s_wait_asynccnt)
#define HAVE_ASYNC_LDS 1
#else
#define HAVE_ASYNC_LDS 0
#endif

#if HAVE_ASYNC_LDS
__device__ __forceinline__ void async_copy16(const void* gsrc, void* ldst) {
    __builtin_amdgcn_global_load_async_to_lds_b128(
        (__attribute__((address_space(1))) v4i*)gsrc,
        (__attribute__((address_space(3))) v4i*)ldst, 0, 0);
}
#endif

__device__ __forceinline__ unsigned short f2bf_bits(float f) {
    union { __bf16 h; unsigned short u; } c;
    c.h = (__bf16)f;            // RNE conversion, native v_cvt on gfx1250
    return c.u;
}

__device__ __forceinline__ void pack8_diff(__bf16* h, const float* a, const float* b) {
    float4 a0 = *(const float4*)a, a1 = *(const float4*)(a + 4);
    float4 b0 = *(const float4*)b, b1 = *(const float4*)(b + 4);
    h[0] = (__bf16)(a0.x - b0.x); h[1] = (__bf16)(a0.y - b0.y);
    h[2] = (__bf16)(a0.z - b0.z); h[3] = (__bf16)(a0.w - b0.w);
    h[4] = (__bf16)(a1.x - b1.x); h[5] = (__bf16)(a1.y - b1.y);
    h[6] = (__bf16)(a1.z - b1.z); h[7] = (__bf16)(a1.w - b1.w);
}

__device__ __forceinline__ void pack8_copy(__bf16* h, const float* a) {
    float4 a0 = *(const float4*)a, a1 = *(const float4*)(a + 4);
    h[0] = (__bf16)a0.x; h[1] = (__bf16)a0.y; h[2] = (__bf16)a0.z; h[3] = (__bf16)a0.w;
    h[4] = (__bf16)a1.x; h[5] = (__bf16)a1.y; h[6] = (__bf16)a1.z; h[7] = (__bf16)a1.w;
}

// ---------------------------------------------------------------------------
// One-shot weight conversion f32 -> bf16 into workspace.
// Layout: [0,32768) W2, [32768,65536) W3, [65536,98304) W4, [98304,163840) W5
// ---------------------------------------------------------------------------
__global__ void __launch_bounds__(256) wcvt_kernel(const float* __restrict__ W2,
                                                   const float* __restrict__ W3,
                                                   const float* __restrict__ W4,
                                                   const float* __restrict__ W5,
                                                   unsigned short* __restrict__ wbf) {
    const int t = blockIdx.x * blockDim.x + threadIdx.x;   // 163840 total
    if (t < 32768)       wbf[t] = f2bf_bits(W2[t]);
    else if (t < 65536)  wbf[t] = f2bf_bits(W3[t - 32768]);
    else if (t < 98304)  wbf[t] = f2bf_bits(W4[t - 65536]);
    else                 wbf[t] = f2bf_bits(W5[t - 98304]);
}

// ---------------------------------------------------------------------------
// KNN: one thread per query point, 256-candidate LDS chunks, register top-16.
// Score s = |xj|^2 - 2 xi.xj  (row-constant |xi|^2 dropped; same ordering).
// ---------------------------------------------------------------------------
__global__ void __launch_bounds__(256) knn_kernel(const float* __restrict__ p,
                                                  int* __restrict__ knn) {
    const int b = blockIdx.x >> 4;                 // 16 blocks per batch
    const int n = ((blockIdx.x & 15) << 8) + threadIdx.x;
    const float* pb = p + (size_t)b * NPTS * 3;
    const float* me = pb + (size_t)n * 3;
    const float px = me[0], py = me[1], pz = me[2];

    __shared__ float sx[256], sy[256], sz[256], sq[256];

    float bd[KNN_K]; int bi[KNN_K];
#pragma unroll
    for (int k = 0; k < KNN_K; ++k) { bd[k] = 3.0e38f; bi[k] = 0x7fffffff; }

    for (int c0 = 0; c0 < NPTS; c0 += 256) {
        __syncthreads();
        {
            const float* q = pb + (size_t)(c0 + threadIdx.x) * 3;
            float x = q[0], y = q[1], z = q[2];
            sx[threadIdx.x] = x; sy[threadIdx.x] = y; sz[threadIdx.x] = z;
            sq[threadIdx.x] = x * x + y * y + z * z;
        }
        __syncthreads();
        for (int jj = 0; jj < 256; ++jj) {
            float s = sq[jj] - 2.0f * (px * sx[jj] + py * sy[jj] + pz * sz[jj]);
            int  ji = c0 + jj;
            if (s < bd[KNN_K - 1] || (s == bd[KNN_K - 1] && ji < bi[KNN_K - 1])) {
                bd[KNN_K - 1] = s; bi[KNN_K - 1] = ji;
#pragma unroll
                for (int t = KNN_K - 1; t > 0; --t) {
                    bool sw = (bd[t] < bd[t - 1]) ||
                              (bd[t] == bd[t - 1] && bi[t] < bi[t - 1]);
                    if (sw) {
                        float td = bd[t]; bd[t] = bd[t - 1]; bd[t - 1] = td;
                        int   ti = bi[t]; bi[t] = bi[t - 1]; bi[t - 1] = ti;
                    }
                }
            }
        }
    }
    int* od = knn + ((size_t)b * NPTS + n) * KNN_K;
#pragma unroll
    for (int k = 0; k < KNN_K; ++k) od[k] = bi[k];
}

// ---------------------------------------------------------------------------
// Edge conv layer 1 (feature dim 6 -> 128): tiny K-dim, plain VALU.
// ---------------------------------------------------------------------------
__global__ void __launch_bounds__(256) edgeconv1_kernel(const float* __restrict__ p,
                                                        const float* __restrict__ W1,
                                                        const int* __restrict__ knn,
                                                        float* __restrict__ xcat) {
    const int t  = blockIdx.x * blockDim.x + threadIdx.x;   // B*N*128
    const int o  = t & 127;
    const int pn = t >> 7;
    const int b  = pn >> 12;
    const float* pp = p + (size_t)pn * 3;
    const float px = pp[0], py = pp[1], pz = pp[2];
    const float* w = W1 + o * 6;
    const float w0 = w[0], w1 = w[1], w2 = w[2];
    const float base = px * w[3] + py * w[4] + pz * w[5];
    float s = 0.0f;
#pragma unroll 4
    for (int k = 0; k < KNN_K; ++k) {
        int j = knn[(size_t)pn * KNN_K + k];
        const float* q = p + ((size_t)b * NPTS + j) * 3;
        float g = (q[0] - px) * w0 + (q[1] - py) * w1 + (q[2] - pz) * w2 + base;
        s += (g > 0.0f) ? g : 0.2f * g;
    }
    xcat[(size_t)pn * 512 + o] = s * (1.0f / 16.0f);
}

// ---------------------------------------------------------------------------
// Edge conv layers 2-4 (Cin=128, feat=256, H=128) on bf16 WMMA.
// Wave per point: A = 16 edges x 256 feat (gathered xj - xi || xi),
// B = bf16 W^T staged in LDS with a 264-ushort padded row stride so the
// per-WMMA ds_load_b128 pattern is bank-conflict free (2-way floor).
// ---------------------------------------------------------------------------
__global__ void __launch_bounds__(256) edgeconv_wmma_kernel(const float* __restrict__ xin,        // row stride 512
                                                            const unsigned short* __restrict__ Wbf, // (128,256) bf16
                                                            const int* __restrict__ knn,
                                                            float* __restrict__ xout) {           // row stride 512
    __shared__ unsigned short ldsW[HDIM * WLD];   // 67.5 KB (padded)

    // Stage bf16 weights: 128 rows x 32 chunks of 16B.
    for (int t = threadIdx.x; t < HDIM * 32; t += 256) {
        const int row = t >> 5, ch = t & 31;
        const unsigned short* src = Wbf + row * 256 + ch * 8;
        unsigned short* dst = &ldsW[row * WLD + ch * 8];
#if HAVE_ASYNC_LDS
        async_copy16(src, dst);
#else
        *(uint4*)dst = *(const uint4*)src;
#endif
    }
#if HAVE_ASYNC_LDS
    __builtin_amdgcn_s_wait_asynccnt(0);
#endif
    __syncthreads();

    const int wid  = threadIdx.x >> 5;
    const int lane = threadIdx.x & 31;
    const int m    = lane & 15;       // edge row / output column
    const int half = lane >> 4;

    const int blocksPerBatch = NPTS / 32;                    // 32 points per block
    const int b     = blockIdx.x / blocksPerBatch;
    const int nbase = (blockIdx.x % blocksPerBatch) * 32 + wid * 4;

    const v8f vzero = {0.f, 0.f, 0.f, 0.f, 0.f, 0.f, 0.f, 0.f};

    for (int pi = 0; pi < 4; ++pi) {
        const int    n    = nbase + pi;
        const size_t rowN = (size_t)b * NPTS + n;
        const int    j    = knn[rowN * KNN_K + m];
        const float* xj   = xin + ((size_t)b * NPTS + j) * 512;
        const float* xn   = xin + rowN * 512;

        v8f acc[8];
#pragma unroll
        for (int ot = 0; ot < 8; ++ot) acc[ot] = vzero;

#pragma unroll
        for (int kt = 0; kt < 8; ++kt) {
            ABu a;
            const int kb = 32 * kt;
            const int c1 = kb + 8 * half;
            if (kb < 128) {                       // xj - xi half of feat
                pack8_diff(a.h,     xj + c1,      xn + c1);
                pack8_diff(a.h + 8, xj + c1 + 16, xn + c1 + 16);
            } else {                              // xi half of feat
                pack8_copy(a.h,     xn + c1 - 128);
                pack8_copy(a.h + 8, xn + c1 - 112);
            }
            const int cb = kb + 8 * half;
#pragma unroll
            for (int ot = 0; ot < 8; ++ot) {
                ABu bt;
                const unsigned short* wrow = &ldsW[(ot * 16 + m) * WLD + cb];
                bt.q[0] = *(const uint4*)(wrow);
                bt.q[1] = *(const uint4*)(wrow + 16);
                acc[ot] = __builtin_amdgcn_wmma_f32_16x16x32_bf16(
                    false, a.v, false, bt.v, (short)0, acc[ot], false, false);
            }
        }

        // LeakyReLU then mean over the 16 edge rows of the D tile.
#pragma unroll
        for (int ot = 0; ot < 8; ++ot) {
            float s = 0.0f;
#pragma unroll
            for (int r = 0; r < 8; ++r) {
                float g = acc[ot][r];
                s += (g > 0.0f) ? g : 0.2f * g;
            }
            s += __shfl_xor(s, 16, 32);           // partner lane holds other 8 edges
            if (lane < 16) xout[rowN * 512 + ot * 16 + lane] = s * (1.0f / 16.0f);
        }
    }
}

// ---------------------------------------------------------------------------
// Final projection: c = lrelu(xcat(16384 x 512) @ W5^T(512 x 128)), bf16 WMMA.
// B-operands are pre-converted bf16: straight 16B loads, zero conversion math.
// ---------------------------------------------------------------------------
__global__ void __launch_bounds__(256) proj_wmma_kernel(const float* __restrict__ xcat,
                                                        const unsigned short* __restrict__ W5bf, // (128,512) bf16
                                                        float* __restrict__ cfeat) {             // (B*N,128)
    const int wid  = threadIdx.x >> 5;
    const int lane = threadIdx.x & 31;
    const int m    = lane & 15;
    const int half = lane >> 4;

    const int tile = blockIdx.x * 8 + wid;        // 0..1023
    const int b    = tile >> 8;
    const int n0   = (tile & 255) * 16;
    const float* arow = xcat + ((size_t)b * NPTS + n0 + m) * 512;

    const v8f vzero = {0.f, 0.f, 0.f, 0.f, 0.f, 0.f, 0.f, 0.f};
    v8f acc[8];
#pragma unroll
    for (int ot = 0; ot < 8; ++ot) acc[ot] = vzero;

#pragma unroll
    for (int kt = 0; kt < 16; ++kt) {
        ABu a;
        const int c1 = 32 * kt + 8 * half;
        pack8_copy(a.h,     arow + c1);
        pack8_copy(a.h + 8, arow + c1 + 16);
        const int cb = 32 * kt + 8 * half;
#pragma unroll
        for (int ot = 0; ot < 8; ++ot) {
            ABu bt;
            const unsigned short* wr = W5bf + (size_t)(ot * 16 + m) * 512 + cb;
            bt.q[0] = *(const uint4*)(wr);
            bt.q[1] = *(const uint4*)(wr + 16);
            acc[ot] = __builtin_amdgcn_wmma_f32_16x16x32_bf16(
                false, a.v, false, bt.v, (short)0, acc[ot], false, false);
        }
    }

#pragma unroll
    for (int ot = 0; ot < 8; ++ot) {
#pragma unroll
        for (int r = 0; r < 8; ++r) {
            float g = acc[ot][r];
            g = (g > 0.0f) ? g : 0.2f * g;
            cfeat[((size_t)b * NPTS + n0 + r + 8 * half) * HDIM + ot * 16 + m] = g;
        }
    }
}

// ---------------------------------------------------------------------------
// Plane features: zero, scatter-add (f32 atomics), divide + transpose.
// ---------------------------------------------------------------------------
__global__ void zero_kernel(float* __restrict__ ptr, long n) {
    long i = (long)blockIdx.x * blockDim.x + threadIdx.x;
    long stride = (long)gridDim.x * blockDim.x;
    for (; i < n; i += stride) ptr[i] = 0.0f;
}

__global__ void __launch_bounds__(256) scatter_kernel(const float* __restrict__ p,
                                                      const float* __restrict__ cfeat,
                                                      float* __restrict__ sums,
                                                      float* __restrict__ cnts) {
    const int t  = blockIdx.x * blockDim.x + threadIdx.x;   // B*N*128
    const int c  = t & 127;
    const int pn = t >> 7;
    const int b  = pn >> 12;
    const float val = cfeat[(size_t)pn * HDIM + c];
    const float* pp = p + (size_t)pn * 3;
    const float q[3] = {pp[0], pp[1], pp[2]};
    const int ca[3]  = {0, 0, 1};                           // xz, xy, yz
    const int cbp[3] = {2, 1, 2};
#pragma unroll
    for (int pl = 0; pl < 3; ++pl) {
        float u = q[ca[pl]]  / 1.101f + 0.5f;
        float v = q[cbp[pl]] / 1.101f + 0.5f;
        u = fminf(fmaxf(u, 0.0f), 1.0f - 1e-5f);
        v = fminf(fmaxf(v, 0.0f), 1.0f - 1e-5f);
        int ix = (int)(u * 64.0f), iy = (int)(v * 64.0f);
        int cell = ix + 64 * iy;
        atomicAdd(&sums[(((size_t)pl * BATCH + b) * 4096 + cell) * HDIM + c], val);
        if (c == 0) atomicAdd(&cnts[((size_t)pl * BATCH + b) * 4096 + cell], 1.0f);
    }
}

__global__ void __launch_bounds__(256) finalize_kernel(const float* __restrict__ sums,
                                                       const float* __restrict__ cnts,
                                                       float* __restrict__ out) {
    const int t    = blockIdx.x * blockDim.x + threadIdx.x; // 3*B*4096*128
    const int c    = t & 127;
    const int rest = t >> 7;
    const int cell = rest & 4095;
    const int pb   = rest >> 12;                            // plane*B + b
    const float cnt = cnts[(size_t)pb * 4096 + cell];
    const float v = sums[((size_t)pb * 4096 + cell) * HDIM + c] / fmaxf(cnt, 1.0f);
    const int pl = pb >> 2, b = pb & 3;
    out[(size_t)pl * (BATCH * HDIM * 4096) + ((size_t)(b * HDIM + c)) * 4096 + cell] = v;
}

// ---------------------------------------------------------------------------
extern "C" void kernel_launch(void* const* d_in, const int* in_sizes, int n_in,
                              void* d_out, int out_size, void* d_ws, size_t ws_size,
                              hipStream_t stream) {
    const float* p  = (const float*)d_in[0];
    const float* W1 = (const float*)d_in[1];
    const float* W2 = (const float*)d_in[2];
    const float* W3 = (const float*)d_in[3];
    const float* W4 = (const float*)d_in[4];
    const float* W5 = (const float*)d_in[5];
    float* out = (float*)d_out;

    // Workspace carve-up (~65.5 MiB total):
    char*  ws    = (char*)d_ws;
    int*   knn   = (int*)ws;                                 //  1.0 MiB
    float* xcat  = (float*)(ws + (1u << 20));                // 32.0 MiB (B,N,512)
    float* cfeat = xcat + (size_t)BATCH * NPTS * 512;        //  8.0 MiB (B,N,128)
    float* sums  = cfeat + (size_t)BATCH * NPTS * HDIM;      // 24.0 MiB (3,B,4096,128)
    float* cnts  = sums + (size_t)3 * BATCH * 4096 * HDIM;   //  0.2 MiB (3,B,4096)
    unsigned short* wbf = (unsigned short*)(cnts + 3 * BATCH * 4096);  // 0.31 MiB bf16 weights

    wcvt_kernel<<<640, 256, 0, stream>>>(W2, W3, W4, W5, wbf);
    knn_kernel<<<BATCH * (NPTS / 256), 256, 0, stream>>>(p, knn);
    edgeconv1_kernel<<<(BATCH * NPTS * HDIM) / 256, 256, 0, stream>>>(p, W1, knn, xcat);
    edgeconv_wmma_kernel<<<BATCH * (NPTS / 32), 256, 0, stream>>>(xcat +   0, wbf +     0, knn, xcat + 128);
    edgeconv_wmma_kernel<<<BATCH * (NPTS / 32), 256, 0, stream>>>(xcat + 128, wbf + 32768, knn, xcat + 256);
    edgeconv_wmma_kernel<<<BATCH * (NPTS / 32), 256, 0, stream>>>(xcat + 256, wbf + 65536, knn, xcat + 384);
    proj_wmma_kernel<<<(BATCH * NPTS / 16) / 8, 256, 0, stream>>>(xcat, wbf + 98304, cfeat);
    zero_kernel<<<2048, 256, 0, stream>>>(sums, (long)3 * BATCH * 4096 * HDIM + 3 * BATCH * 4096);
    scatter_kernel<<<(BATCH * NPTS * HDIM) / 256, 256, 0, stream>>>(p, cfeat, sums, cnts);
    finalize_kernel<<<(3 * BATCH * NPTS * HDIM) / 256, 256, 0, stream>>>(sums, cnts, out);
}
